// MultiHeadAttention_61168924230307
// MI455X (gfx1250) — compile-verified, matrix operand order checked by reading, not run on hardware
//
#include <hip/hip_runtime.h>

// ---------------- problem constants ----------------
#define BB   2
#define SS   2048
#define EE   1024
#define HH   16
#define HD   64
#define E3   3072           // 3*E
#define MM   (BB*SS)        // 4096 tokens

typedef __attribute__((ext_vector_type(16))) __bf16 v16bf;
typedef __attribute__((ext_vector_type(2)))  __bf16 v2bf;
typedef __attribute__((ext_vector_type(8)))  float  v8f;
typedef __attribute__((ext_vector_type(4)))  unsigned int u32x4;
typedef __attribute__((ext_vector_type(8)))  int i32x8;
typedef __attribute__((ext_vector_type(4)))  int i32x4;

__device__ __forceinline__ v8f wmma_bf16(v16bf a, v16bf b, v8f c) {
  // (neg_a, A, neg_b, B, c_mod, C, reuse_a, reuse_b)
  return __builtin_amdgcn_wmma_f32_16x16x32_bf16(false, a, false, b, (short)0, c,
                                                 false, false);
}

// dword offset (within a 16-dword K-chunk row) holding the bf16 pair for
// fragment slot v (0..7) of a 16-bit A/B matrix operand; hl = lane>>4.
__device__ __forceinline__ int fragDword(int v, int hl) {
  return (v & 3) + ((v >> 2) << 3) + (hl << 2);
}

__device__ __forceinline__ v8f zero8() {
  v8f z;
#pragma unroll
  for (int i = 0; i < 8; ++i) z[i] = 0.0f;
  return z;
}

// LDS byte offset of a __shared__ object (generic ptr low 32 bits = LDS offset)
__device__ __forceinline__ unsigned int lds_offset(const void* p) {
  return (unsigned int)(unsigned long long)(uintptr_t)p;
}

// ---- Tensor Data Mover: 2D tile (rows x cols bf16) global -> LDS ----------
// Pads every 16 LDS dwords with 4 dwords => LDS row stride 40 bf16 (LROW).
// D# per CDNA5 ISA ch.8: group0 {count,lds,global,type}, group1 {sizes/strides}.
__device__ __forceinline__ void tdm_load_tile_bf16(
    const __bf16* gptr,        // global address of tile start
    unsigned int lds_byte_off, // LDS destination byte offset
    unsigned int rows,         // tile_dim1 (tile rows)
    unsigned int cols,         // tile_dim0 (bf16 elements per row)
    unsigned int gstride,      // tensor row stride, elements (== tensor_dim0)
    unsigned int tensor_rows)  // tensor_dim1 (for OOB checks)
{
  const unsigned long long ga = (unsigned long long)(uintptr_t)gptr;
  u32x4 g0;
  g0[0] = 1u;                                   // count=1, user descriptor
  g0[1] = lds_byte_off;                         // lds_addr
  g0[2] = (unsigned int)(ga & 0xffffffffull);   // global_addr[31:0]
  g0[3] = (unsigned int)((ga >> 32) & 0x01ffffffull) | (2u << 30); // [56:32]|type=2

  i32x8 g1;
  const unsigned int data_size   = 1u;  // 2-byte elements
  const unsigned int pad_enable  = 1u;
  const unsigned int pad_interval= 3u;  // every 16 dwords (64B of data)
  const unsigned int pad_amount  = 3u;  // insert 4 dwords (16B) of pad
  g1[0] = (int)((data_size << 16) | (pad_enable << 20) |
                (pad_interval << 22) | (pad_amount << 25)); // wg_mask=0
  g1[1] = (int)((gstride & 0xffffu) << 16);                 // tensor_dim0 lo16
  g1[2] = (int)((gstride >> 16) | ((tensor_rows & 0xffffu) << 16));
  g1[3] = (int)((tensor_rows >> 16) | (cols << 16));        // tile_dim0
  g1[4] = (int)(rows);                                      // tile_dim1 (dim2=0)
  g1[5] = (int)(gstride);                                   // dim0_stride lo32
  g1[6] = 0;                                                // dim0_stride hi=0
  g1[7] = 0;

  i32x4 gz4;
  gz4[0] = 0; gz4[1] = 0; gz4[2] = 0; gz4[3] = 0;           // 2D: groups 2/3 off
  i32x8 gz8;
#pragma unroll
  for (int i = 0; i < 8; ++i) gz8[i] = 0;
  // 6-arg toolchain form: (g0 u32x4, g1 i32x8, g2 i32x4, g3 i32x4, i32x8, cpol)
  __builtin_amdgcn_tensor_load_to_lds(g0, g1, gz4, gz4, gz8, 0);
}

// ---------------- f32 -> bf16 conversion ----------------
__global__ void cvt_f32_bf16(const float* __restrict__ in,
                             __bf16* __restrict__ out, int n) {
  int i = blockIdx.x * blockDim.x + threadIdx.x;
  int stride = gridDim.x * blockDim.x;
  for (; i < n; i += stride) out[i] = (__bf16)in[i];
}

// ---------------- bf16 WMMA GEMM:  C[M,N] = A[M,K] * W[N,K]^T + bias ------
// block = 256 threads = 8 waves; tile 128x128, BK=32; TDM-fed double buffer.
#define BM 128
#define BN 128
#define BK 32
#define LROW 40   // bf16 elements per LDS row (20 dwords: 16 data + 4 pad)

template <bool OUT_BF16>
__global__ __launch_bounds__(256)
void gemm_bf16_wmma(const __bf16* __restrict__ A, const __bf16* __restrict__ W,
                    const float* __restrict__ bias, void* __restrict__ Cout,
                    int M, int N, int K) {
  __shared__ __align__(16) __bf16 As[2][BM * LROW];
  __shared__ __align__(16) __bf16 Bs[2][BN * LROW];

  const int tid  = threadIdx.x;
  const int wave = tid >> 5;
  const int lane = tid & 31;
  const int hl   = lane >> 4;
  const int mloc = lane & 15;
  const int wm   = (wave >> 2) * 64;   // 2 waves along M
  const int wn   = (wave & 3) * 32;    // 4 waves along N
  const int m0   = blockIdx.y * BM;
  const int n0   = blockIdx.x * BN;

  v8f acc[4][2];
#pragma unroll
  for (int nt = 0; nt < 2; ++nt) {
    const float bv = bias[n0 + wn + nt * 16 + mloc];
#pragma unroll
    for (int mt = 0; mt < 4; ++mt) {
#pragma unroll
      for (int r = 0; r < 8; ++r) acc[mt][nt][r] = bv;   // bias depends on N only
    }
  }

  // ---- TDM prologue: fetch first tiles into buffer 0 ----
  if (wave == 0) {
    tdm_load_tile_bf16(A + (size_t)m0 * K, lds_offset(&As[0][0]), BM, BK, K, M);
    tdm_load_tile_bf16(W + (size_t)n0 * K, lds_offset(&Bs[0][0]), BN, BK, K, N);
  }

  for (int k0 = 0, it = 0; k0 < K; k0 += BK, ++it) {
    const int cur = it & 1;
    const bool more = (k0 + BK) < K;
    if (wave == 0) {
      if (more) {  // prefetch next tile into the other buffer
        tdm_load_tile_bf16(A + (size_t)m0 * K + (k0 + BK),
                           lds_offset(&As[cur ^ 1][0]), BM, BK, K, M);
        tdm_load_tile_bf16(W + (size_t)n0 * K + (k0 + BK),
                           lds_offset(&Bs[cur ^ 1][0]), BN, BK, K, N);
        __builtin_amdgcn_s_wait_tensorcnt(2);  // current buffer complete
      } else {
        __builtin_amdgcn_s_wait_tensorcnt(0);
      }
    }
    __syncthreads();   // current tiles visible to all waves

    // ---- fragments from LDS ----
    v16bf af[4];
#pragma unroll
    for (int mt = 0; mt < 4; ++mt) {
      const int row = wm + mt * 16 + mloc;
#pragma unroll
      for (int v = 0; v < 8; ++v) {
        const int d = fragDword(v, hl);
        const v2bf p = *reinterpret_cast<const v2bf*>(&As[cur][row * LROW + 2 * d]);
        af[mt][2 * v] = p[0];
        af[mt][2 * v + 1] = p[1];
      }
    }
    v16bf bfr[2];
#pragma unroll
    for (int nt = 0; nt < 2; ++nt) {
      const int row = wn + nt * 16 + mloc;
#pragma unroll
      for (int v = 0; v < 8; ++v) {
        const int d = fragDword(v, hl);
        const v2bf p = *reinterpret_cast<const v2bf*>(&Bs[cur][row * LROW + 2 * d]);
        bfr[nt][2 * v] = p[0];
        bfr[nt][2 * v + 1] = p[1];
      }
    }

#pragma unroll
    for (int mt = 0; mt < 4; ++mt)
#pragma unroll
      for (int nt = 0; nt < 2; ++nt)
        acc[mt][nt] = wmma_bf16(af[mt], bfr[nt], acc[mt][nt]);

    __syncthreads();   // all reads done before this buffer is re-filled
  }

  // ---- epilogue: C layout = (row r + 8*hl, col = lane&15) per VGPR r ----
#pragma unroll
  for (int mt = 0; mt < 4; ++mt) {
#pragma unroll
    for (int nt = 0; nt < 2; ++nt) {
#pragma unroll
      for (int r = 0; r < 8; ++r) {
        const int row = m0 + wm + mt * 16 + r + 8 * hl;
        const int col = n0 + wn + nt * 16 + mloc;
        if (OUT_BF16)
          ((__bf16*)Cout)[(size_t)row * N + col] = (__bf16)acc[mt][nt][r];
        else
          ((float*)Cout)[(size_t)row * N + col] = acc[mt][nt][r];
      }
    }
  }
}

// ---------------- flash attention over qkv (bf16), faithful mask ----------
// qkv: [B*S, 3E] row-major; token row = [H x (q[64] k[64] v[64])]
// One wave handles one (b,h) 16-query tile over all S keys (32 per chunk).
__global__ __launch_bounds__(128)
void flash_attn(const __bf16* __restrict__ qkv, __bf16* __restrict__ outv) {
  __shared__ __align__(16) __bf16 pbuf[4][16 * LROW];

  const int lane = threadIdx.x & 31;
  const int wave = threadIdx.x >> 5;
  const int hl   = lane >> 4;
  const int mloc = lane & 15;

  const int QT   = SS / 16;                 // 128 q-tiles per (b,h)
  const int tile = blockIdx.x * 4 + wave;   // 0..4095
  const int b    = tile / (HH * QT);
  const int rem  = tile % (HH * QT);
  const int h    = rem / QT;
  const int qt   = rem % QT;
  const int qbase = qt * 16;

  const __bf16* base = qkv + (size_t)b * SS * E3 + (size_t)h * (3 * HD);

  // Q fragments (16x64 A-operand => two 16x32 chunks)
  v16bf aq0, aq1;
  {
    const __bf16* qrow = base + (size_t)(qbase + mloc) * E3;  // q section = +0
#pragma unroll
    for (int v = 0; v < 8; ++v) {
      const int d = fragDword(v, hl);
      const v2bf p0 = *reinterpret_cast<const v2bf*>(qrow + 2 * d);
      const v2bf p1 = *reinterpret_cast<const v2bf*>(qrow + 32 + 2 * d);
      aq0[2 * v] = p0[0]; aq0[2 * v + 1] = p0[1];
      aq1[2 * v] = p1[0]; aq1[2 * v + 1] = p1[1];
    }
  }

  v8f o[4];
#pragma unroll
  for (int dt = 0; dt < 4; ++dt) o[dt] = zero8();
  float mrow[8], lrow[8];
#pragma unroll
  for (int r = 0; r < 8; ++r) { mrow[r] = -3.0e38f; lrow[r] = 0.0f; }

  const float scale = 0.125f;               // 1/sqrt(64)

  for (int kc = 0; kc < SS; kc += 32) {
    // ---- S = Q K^T for a 16x32 key chunk (two 16x16 n-tiles) ----
    v8f s[2];
#pragma unroll
    for (int nt = 0; nt < 2; ++nt) {
      const __bf16* krow = base + (size_t)(kc + nt * 16 + mloc) * E3 + HD; // k sec
      v16bf bk0, bk1;
#pragma unroll
      for (int v = 0; v < 8; ++v) {
        const int d = fragDword(v, hl);
        const v2bf p0 = *reinterpret_cast<const v2bf*>(krow + 2 * d);
        const v2bf p1 = *reinterpret_cast<const v2bf*>(krow + 32 + 2 * d);
        bk0[2 * v] = p0[0]; bk0[2 * v + 1] = p0[1];
        bk1[2 * v] = p1[0]; bk1[2 * v + 1] = p1[1];
      }
      v8f c = zero8();
      c = wmma_bf16(aq0, bk0, c);
      c = wmma_bf16(aq1, bk1, c);
      s[nt] = c;
    }

    // ---- scale + faithful mask (k <= q gets -1e9), online softmax ----
#pragma unroll
    for (int r = 0; r < 8; ++r) {
      const int qg  = qbase + r + 8 * hl;
      const int kg0 = kc + mloc;
      const int kg1 = kc + 16 + mloc;
      float sv0 = s[0][r] * scale + ((kg0 <= qg) ? -1.0e9f : 0.0f);
      float sv1 = s[1][r] * scale + ((kg1 <= qg) ? -1.0e9f : 0.0f);

      float t = fmaxf(sv0, sv1);
#pragma unroll
      for (int m = 8; m >= 1; m >>= 1) t = fmaxf(t, __shfl_xor(t, m, 32));
      const float mnew  = fmaxf(mrow[r], t);
      const float alpha = __expf(mrow[r] - mnew);
      mrow[r] = mnew;

      const float p0 = __expf(sv0 - mnew);
      const float p1 = __expf(sv1 - mnew);
      float rs = p0 + p1;
#pragma unroll
      for (int m = 8; m >= 1; m >>= 1) rs += __shfl_xor(rs, m, 32);
      lrow[r] = lrow[r] * alpha + rs;

#pragma unroll
      for (int dt = 0; dt < 4; ++dt) o[dt][r] *= alpha;

      // stash P (C-layout -> LDS) for re-read in A-layout
      pbuf[wave][(r + 8 * hl) * LROW + mloc]      = (__bf16)p0;
      pbuf[wave][(r + 8 * hl) * LROW + 16 + mloc] = (__bf16)p1;
    }
    __syncthreads();

    // ---- P (16x32) as A-operand ----
    v16bf pa;
#pragma unroll
    for (int v = 0; v < 8; ++v) {
      const int d = fragDword(v, hl);
      const v2bf p = *reinterpret_cast<const v2bf*>(&pbuf[wave][mloc * LROW + 2 * d]);
      pa[2 * v] = p[0]; pa[2 * v + 1] = p[1];
    }

    // ---- O += P V : four 16-wide d-tiles ----
#pragma unroll
    for (int dt = 0; dt < 4; ++dt) {
      v16bf bv;
#pragma unroll
      for (int v = 0; v < 8; ++v) {
        const int kb = 2 * fragDword(v, hl);   // key pair (kb, kb+1) in chunk
        const __bf16* vrow =
            base + (size_t)(kc + kb) * E3 + 2 * HD + dt * 16 + mloc;  // v sec
        bv[2 * v]     = vrow[0];
        bv[2 * v + 1] = vrow[E3];
      }
      o[dt] = wmma_bf16(pa, bv, o[dt]);
    }
    __syncthreads();
  }

  // ---- normalize + write values layout [B*S, E] (bf16) ----
#pragma unroll
  for (int r = 0; r < 8; ++r) {
    const float inv = 1.0f / lrow[r];
    const size_t row = (size_t)b * SS + qbase + r + 8 * hl;
#pragma unroll
    for (int dt = 0; dt < 4; ++dt) {
      outv[row * EE + h * HD + dt * 16 + mloc] = (__bf16)(o[dt][r] * inv);
    }
  }
}

// ---------------- launcher ----------------
extern "C" void kernel_launch(void* const* d_in, const int* in_sizes, int n_in,
                              void* d_out, int out_size, void* d_ws, size_t ws_size,
                              hipStream_t stream) {
  const float* x    = (const float*)d_in[0];   // [B,S,E]
  const float* Wqkv = (const float*)d_in[1];   // [3E,E]
  const float* bqkv = (const float*)d_in[2];   // [3E]
  const float* Wout = (const float*)d_in[3];   // [E,E]
  const float* bout = (const float*)d_in[4];   // [E]

  char* ws = (char*)d_ws;
  __bf16* xb    = (__bf16*)(ws);                         //  8 MiB
  __bf16* wqb   = (__bf16*)(ws + 8388608);               //  6 MiB
  __bf16* wob   = (__bf16*)(ws + 14680064);              //  2 MiB
  __bf16* qkvb  = (__bf16*)(ws + 16777216);              // 24 MiB
  __bf16* attnb = (__bf16*)(ws + 41943040);              //  8 MiB  (total 48 MiB)

  // 1) f32 -> bf16 conversions
  cvt_f32_bf16<<<2048, 256, 0, stream>>>(x,    xb,  MM * EE);
  cvt_f32_bf16<<<2048, 256, 0, stream>>>(Wqkv, wqb, E3 * EE);
  cvt_f32_bf16<<<1024, 256, 0, stream>>>(Wout, wob, EE * EE);

  // 2) qkv = x @ Wqkv^T + bqkv   (bf16 out)
  {
    dim3 grid(E3 / BN, MM / BM);   // 24 x 32
    gemm_bf16_wmma<true><<<grid, 256, 0, stream>>>(xb, wqb, bqkv, (void*)qkvb,
                                                   MM, E3, EE);
  }

  // 3) attention (flash, faithful tril-incl-diagonal mask)
  flash_attn<<<(BB * HH * (SS / 16)) / 4, 128, 0, stream>>>(qkvb, attnb);

  // 4) out = values @ Wout^T + bout  (f32 out)
  {
    dim3 grid(EE / BN, MM / BM);   // 8 x 32
    gemm_bf16_wmma<false><<<grid, 256, 0, stream>>>(attnb, wob, bout, d_out,
                                                    MM, EE, EE);
  }
  (void)in_sizes; (void)n_in; (void)out_size; (void)ws_size;
}
